// CrossAttention_22247930593740
// MI455X (gfx1250) — matrix operand order repaired and checked by reading, not compiled, not run on hardware
//
#include <hip/hip_runtime.h>

// ---------------- problem constants ----------------
constexpr int Bn = 2, Nn = 4096, Dn = 320, Hn = 8, DHn = 64, INNERn = 512;
constexpr int BHn = Bn * Hn;                       // 16
constexpr int Mrows = Bn * Nn;                     // 8192
constexpr size_t SZ_FO  = (size_t)Bn * Nn * Dn;    // 2,621,440
constexpr size_t SZ_QKV = (size_t)BHn * Nn * DHn;  // 4,194,304
constexpr size_t SZ_W   = (size_t)Dn * INNERn;     // 163,840

// ---------------- types ----------------
typedef __attribute__((ext_vector_type(16))) __bf16        v16bf;
typedef __attribute__((ext_vector_type(8)))  float         v8f;
typedef __attribute__((ext_vector_type(8)))  unsigned int  v8u;
typedef int v4i_vs __attribute__((vector_size(16)));       // b128 payload type
typedef __attribute__((address_space(1))) v4i_vs* gv4i_p;  // global b128 ptr
typedef __attribute__((address_space(3))) v4i_vs* lv4i_p;  // LDS b128 ptr

// ---------------- helpers ----------------
__device__ __forceinline__ unsigned short f2bf(float f) {  // RNE f32->bf16
  unsigned int u = __builtin_bit_cast(unsigned int, f);
  u = (u + 0x7fffu + ((u >> 16) & 1u)) >> 16;
  return (unsigned short)u;
}
__device__ __forceinline__ uint2 f2bf_pack4(float4 f) {
  uint2 r;
  r.x = (unsigned int)f2bf(f.x) | ((unsigned int)f2bf(f.y) << 16);
  r.y = (unsigned int)f2bf(f.z) | ((unsigned int)f2bf(f.w) << 16);
  return r;
}

// Async global->LDS 16-byte copy if the gfx1250 builtin exists; else uint4 copy.
#if defined(__has_builtin)
#if __has_builtin(__builtin_amdgcn_global_load_async_to_lds_b128) && \
    __has_builtin(__builtin_amdgcn_s_wait_asynccnt)
#define HAVE_ASYNC_LDS 1
#endif
#endif
#ifndef HAVE_ASYNC_LDS
#define HAVE_ASYNC_LDS 0
#endif

__device__ __forceinline__ void copy16_g2l(const unsigned short* __restrict__ g,
                                           unsigned short* __restrict__ l) {
#if HAVE_ASYNC_LDS
  __builtin_amdgcn_global_load_async_to_lds_b128((gv4i_p)(void*)g,
                                                 (lv4i_p)(void*)l, 0, 0);
#else
  *(uint4*)l = *(const uint4*)g;
#endif
}
__device__ __forceinline__ void wait_async() {
#if HAVE_ASYNC_LDS
  __builtin_amdgcn_s_wait_asynccnt(0);
#endif
}

// 16x32 bf16 fragment from an LDS tile (row-start pointer). IS_B = B-matrix map.
template <bool IS_B>
__device__ __forceinline__ v16bf ld_frag(const unsigned short* base, int hi) {
  v8u u;
#pragma unroll
  for (int p = 0; p < 8; ++p) {
    int k = 2 * p + (p >= 4 ? 8 : 0) + 8 * hi;
    u[p] = *reinterpret_cast<const unsigned int*>(base + k);
  }
  return __builtin_bit_cast(v16bf, u);
}
__device__ __forceinline__ v8f wmma_bf16(v16bf a, v16bf b, v8f c) {
  return __builtin_amdgcn_wmma_f32_16x16x32_bf16(false, a, false, b,
                                                 (short)0, c, false, false);
}

// ================= prep kernels (bf16 workspace) =================
__global__ __launch_bounds__(256) void conv_f2bf_kernel(
    const float* __restrict__ src, unsigned short* __restrict__ dst, int n4) {
  int i = blockIdx.x * blockDim.x + threadIdx.x;
  if (i < n4) {
    float4 f = reinterpret_cast<const float4*>(src)[i];
    reinterpret_cast<uint2*>(dst)[i] = f2bf_pack4(f);
  }
}
// Wt[n*K + k] = bf16(W[k*Nw + n])
__global__ __launch_bounds__(256) void transpose_w_kernel(
    const float* __restrict__ W, unsigned short* __restrict__ Wt, int K, int Nw) {
  int i = blockIdx.x * blockDim.x + threadIdx.x;
  if (i < K * Nw) {
    int k = i / Nw, n = i % Nw;
    Wt[(size_t)n * K + k] = f2bf(W[i]);
  }
}

// ================= kernel 1: QKV projections =================
// grid (512, 24); block 128 (4 waves); 16 rows x 64 cols per block.
template <bool PRE>
__global__ __launch_bounds__(128) void proj_kernel(
    const float* __restrict__ x, const float* __restrict__ Wq,
    const float* __restrict__ Wk, const float* __restrict__ Wv,
    const unsigned short* __restrict__ xbf, const unsigned short* __restrict__ wqt,
    const unsigned short* __restrict__ wkt, const unsigned short* __restrict__ wvt,
    float* __restrict__ oq, float* __restrict__ ok, float* __restrict__ ov,
    unsigned short* __restrict__ qbf, unsigned short* __restrict__ kbf,
    unsigned short* __restrict__ vbf) {
  __shared__ __align__(16) unsigned short As[16][48];
  __shared__ __align__(16) unsigned short Bs[64][48];
  const int tid = threadIdx.x, wave = tid >> 5, lane = tid & 31;
  const int mloc = lane & 15, hi = lane >> 4;
  const int r0 = blockIdx.x * 16;
  const int wsel = blockIdx.y >> 3;          // 0=q 1=k 2=v
  const int lc0 = (blockIdx.y & 7) * 64;
  const float* W = wsel == 0 ? Wq : (wsel == 1 ? Wk : Wv);
  const unsigned short* Wt = wsel == 0 ? wqt : (wsel == 1 ? wkt : wvt);
  float* outp = wsel == 0 ? oq : (wsel == 1 ? ok : ov);
  unsigned short* outbf = wsel == 0 ? qbf : (wsel == 1 ? kbf : vbf);

  v8f acc = {};
  for (int kc = 0; kc < Dn / 32; ++kc) {
    const int k0 = kc * 32;
    __syncthreads();
    if (PRE) {
      if (tid < 64) {                                   // A: 64 x b128
        int m = tid >> 2, kq = (tid & 3) * 8;
        copy16_g2l(&xbf[(size_t)(r0 + m) * Dn + k0 + kq], &As[m][kq]);
      }
#pragma unroll
      for (int i = 0; i < 2; ++i) {                     // B: 256 x b128
        int e = tid + i * 128;
        int n = e >> 2, kq = (e & 3) * 8;
        copy16_g2l(&Wt[(size_t)(lc0 + n) * Dn + k0 + kq], &Bs[n][kq]);
      }
      wait_async();
    } else {
      {                                                 // A: 128 x float4
        int m = tid >> 3, kq = (tid & 7) * 4;
        float4 f = *(const float4*)&x[(size_t)(r0 + m) * Dn + k0 + kq];
        *(uint2*)&As[m][kq] = f2bf_pack4(f);
      }
#pragma unroll
      for (int i = 0; i < 4; ++i) {                     // B: 512 x float4
        int e = tid + i * 128;
        int k = e >> 4, nq = (e & 15) * 4;
        float4 f = *(const float4*)&W[(size_t)(k0 + k) * INNERn + lc0 + nq];
        Bs[nq + 0][k] = f2bf(f.x); Bs[nq + 1][k] = f2bf(f.y);
        Bs[nq + 2][k] = f2bf(f.z); Bs[nq + 3][k] = f2bf(f.w);
      }
    }
    __syncthreads();
    v16bf a = ld_frag<false>(&As[mloc][0], hi);
    v16bf b = ld_frag<true>(&Bs[wave * 16 + mloc][0], hi);
    acc = wmma_bf16(a, b, acc);
  }
  const int col = lc0 + wave * 16 + mloc;
  const int h = col >> 6, d = col & 63;
#pragma unroll
  for (int r = 0; r < 8; ++r) {
    int g = r0 + r + 8 * hi;
    int bb = g >> 12, nn = g & 4095;
    size_t idx = (((size_t)bb * Hn + h) * Nn + nn) * DHn + d;
    outp[idx] = acc[r];
    if (PRE) outbf[idx] = f2bf(acc[r]);
  }
}

// ================= kernel 2: flash attention =================
// grid (64, 16); block 128 (4 waves, 16 q-rows each).
template <bool PRE>
__global__ __launch_bounds__(128) void attn_kernel(
    const float* __restrict__ q, const float* __restrict__ k,
    const float* __restrict__ v, const unsigned short* __restrict__ qbf,
    const unsigned short* __restrict__ kbf, const unsigned short* __restrict__ vbf,
    float* __restrict__ inter, unsigned short* __restrict__ ibf) {
  __shared__ __align__(16) unsigned short Qs[64][72];
  __shared__ __align__(16) unsigned short Ks[64][72];
  __shared__ __align__(16) unsigned short Vs[64][72];   // transposed: [d][j]
  __shared__ __align__(16) unsigned short Ps[4][16][72];
  const int tid = threadIdx.x, wave = tid >> 5, lane = tid & 31;
  const int mloc = lane & 15, hi = lane >> 4;
  const int q0 = blockIdx.x * 64;
  const size_t base = (size_t)blockIdx.y * Nn * DHn;

  if (PRE) {
#pragma unroll
    for (int i = 0; i < 4; ++i) {
      int e = tid + i * 128;
      int m = e >> 3, dq = (e & 7) * 8;
      copy16_g2l(&qbf[base + (size_t)(q0 + m) * DHn + dq], &Qs[m][dq]);
    }
    wait_async();
  } else {
#pragma unroll
    for (int i = 0; i < 8; ++i) {
      int e = tid + i * 128;
      int m = e >> 4, dq = (e & 15) * 4;
      float4 f = *(const float4*)&q[base + (size_t)(q0 + m) * DHn + dq];
      *(uint2*)&Qs[m][dq] = f2bf_pack4(f);
    }
  }
  __syncthreads();
  const v16bf aq0 = ld_frag<false>(&Qs[wave * 16 + mloc][0], hi);
  const v16bf aq1 = ld_frag<false>(&Qs[wave * 16 + mloc][32], hi);

  v8f o[4] = {v8f{}, v8f{}, v8f{}, v8f{}};
  float mrow[8], lrow[8];
#pragma unroll
  for (int r = 0; r < 8; ++r) { mrow[r] = -3.0e38f; lrow[r] = 0.0f; }
  const float scale = 0.125f;

  for (int kt = 0; kt < Nn / 64; ++kt) {
    const int kb = kt * 64;
    if (PRE) {
#pragma unroll
      for (int i = 0; i < 4; ++i) {                     // K: 512 x b128
        int e = tid + i * 128;
        int j = e >> 3, dq = (e & 7) * 8;
        copy16_g2l(&kbf[base + (size_t)(kb + j) * DHn + dq], &Ks[j][dq]);
      }
#pragma unroll
      for (int i = 0; i < 8; ++i) {                     // V: transpose copies
        int e = tid + i * 128;
        int j = e >> 4, dq = (e & 15) * 4;
        uint2 u = *(const uint2*)&vbf[base + (size_t)(kb + j) * DHn + dq];
        Vs[dq + 0][j] = (unsigned short)(u.x & 0xffffu);
        Vs[dq + 1][j] = (unsigned short)(u.x >> 16);
        Vs[dq + 2][j] = (unsigned short)(u.y & 0xffffu);
        Vs[dq + 3][j] = (unsigned short)(u.y >> 16);
      }
      wait_async();
    } else {
#pragma unroll
      for (int i = 0; i < 8; ++i) {
        int e = tid + i * 128;
        int j = e >> 4, dq = (e & 15) * 4;
        float4 fk = *(const float4*)&k[base + (size_t)(kb + j) * DHn + dq];
        float4 fv = *(const float4*)&v[base + (size_t)(kb + j) * DHn + dq];
        *(uint2*)&Ks[j][dq] = f2bf_pack4(fk);
        Vs[dq + 0][j] = f2bf(fv.x); Vs[dq + 1][j] = f2bf(fv.y);
        Vs[dq + 2][j] = f2bf(fv.z); Vs[dq + 3][j] = f2bf(fv.w);
      }
    }
    if (kt + 1 < Nn / 64) {                             // prefetch next tile
      int j = tid >> 1, sel = (tid & 1) * 32;
      if (PRE) {
        __builtin_prefetch(&kbf[base + (size_t)(kb + 64 + j) * DHn + sel], 0, 1);
        __builtin_prefetch(&vbf[base + (size_t)(kb + 64 + j) * DHn + sel], 0, 1);
      } else {
        __builtin_prefetch(&k[base + (size_t)(kb + 64 + j) * DHn + sel], 0, 1);
        __builtin_prefetch(&v[base + (size_t)(kb + 64 + j) * DHn + sel], 0, 1);
      }
    }
    __syncthreads();

    // S = q @ k^T  (16x64 per wave)
    v8f s[4] = {v8f{}, v8f{}, v8f{}, v8f{}};
#pragma unroll
    for (int t = 0; t < 4; ++t) {
      v16bf b0 = ld_frag<true>(&Ks[t * 16 + mloc][0], hi);
      s[t] = wmma_bf16(aq0, b0, s[t]);
      v16bf b1 = ld_frag<true>(&Ks[t * 16 + mloc][32], hi);
      s[t] = wmma_bf16(aq1, b1, s[t]);
    }
#pragma unroll
    for (int t = 0; t < 4; ++t) s[t] *= scale;

    // online softmax (row = r + 8*hi; rows shared by 16-lane groups)
#pragma unroll
    for (int r = 0; r < 8; ++r) {
      float mt = fmaxf(fmaxf(s[0][r], s[1][r]), fmaxf(s[2][r], s[3][r]));
      mt = fmaxf(mt, __shfl_xor(mt, 1, 32));
      mt = fmaxf(mt, __shfl_xor(mt, 2, 32));
      mt = fmaxf(mt, __shfl_xor(mt, 4, 32));
      mt = fmaxf(mt, __shfl_xor(mt, 8, 32));
      float mnew = fmaxf(mrow[r], mt);
      float alpha = __expf(mrow[r] - mnew);
      float p0 = __expf(s[0][r] - mnew), p1 = __expf(s[1][r] - mnew);
      float p2 = __expf(s[2][r] - mnew), p3 = __expf(s[3][r] - mnew);
      float lt = p0 + p1 + p2 + p3;
      lt += __shfl_xor(lt, 1, 32); lt += __shfl_xor(lt, 2, 32);
      lt += __shfl_xor(lt, 4, 32); lt += __shfl_xor(lt, 8, 32);
      lrow[r] = lrow[r] * alpha + lt;
      mrow[r] = mnew;
      o[0][r] *= alpha; o[1][r] *= alpha; o[2][r] *= alpha; o[3][r] *= alpha;
      const int mr = r + 8 * hi;
      Ps[wave][mr][0 * 16 + mloc] = f2bf(p0);
      Ps[wave][mr][1 * 16 + mloc] = f2bf(p1);
      Ps[wave][mr][2 * 16 + mloc] = f2bf(p2);
      Ps[wave][mr][3 * 16 + mloc] = f2bf(p3);
    }

    // O += P @ V
    v16bf ap0 = ld_frag<false>(&Ps[wave][mloc][0], hi);
    v16bf ap1 = ld_frag<false>(&Ps[wave][mloc][32], hi);
#pragma unroll
    for (int t = 0; t < 4; ++t) {
      v16bf b0 = ld_frag<true>(&Vs[t * 16 + mloc][0], hi);
      o[t] = wmma_bf16(ap0, b0, o[t]);
      v16bf b1 = ld_frag<true>(&Vs[t * 16 + mloc][32], hi);
      o[t] = wmma_bf16(ap1, b1, o[t]);
    }
    __syncthreads();
  }

  float* ip = inter + base;
  unsigned short* ib = ibf + base;
#pragma unroll
  for (int r = 0; r < 8; ++r) {
    float inv = 1.0f / lrow[r];
    int nn = q0 + wave * 16 + r + 8 * hi;
#pragma unroll
    for (int t = 0; t < 4; ++t) {
      float val = o[t][r] * inv;
      size_t ix = (size_t)nn * DHn + t * 16 + mloc;
      ip[ix] = val;
      if (PRE) ib[ix] = f2bf(val);
    }
  }
}

// ================= kernel 3: output projection + bias =================
// grid (512, 5); block 128; 16 rows x 64 cols; K = 512.
template <bool PRE>
__global__ __launch_bounds__(128) void outproj_kernel(
    const float* __restrict__ inter, const unsigned short* __restrict__ ibf,
    const float* __restrict__ Wo, const unsigned short* __restrict__ wot,
    const float* __restrict__ bo, float* __restrict__ outf) {
  __shared__ __align__(16) unsigned short As[16][48];
  __shared__ __align__(16) unsigned short Bs[64][48];
  const int tid = threadIdx.x, wave = tid >> 5, lane = tid & 31;
  const int mloc = lane & 15, hi = lane >> 4;
  const int r0 = blockIdx.x * 16;
  const int c0 = blockIdx.y * 64;
  v8f acc = {};
  for (int kc = 0; kc < INNERn / 32; ++kc) {
    const int k0 = kc * 32;
    __syncthreads();
    if (PRE) {
      if (tid < 64) {                                   // A: 64 x b128
        int m = tid >> 2, kq = (tid & 3) * 8;
        int g = r0 + m, bb = g >> 12, nn = g & 4095;
        int kcol = k0 + kq, h = kcol >> 6, d = kcol & 63;
        copy16_g2l(&ibf[(((size_t)bb * Hn + h) * Nn + nn) * DHn + d], &As[m][kq]);
      }
#pragma unroll
      for (int i = 0; i < 2; ++i) {                     // B: 256 x b128
        int e = tid + i * 128;
        int n = e >> 2, kq = (e & 3) * 8;
        copy16_g2l(&wot[(size_t)(c0 + n) * INNERn + k0 + kq], &Bs[n][kq]);
      }
      wait_async();
    } else {
      {
        int m = tid >> 3, kq = (tid & 7) * 4;
        int g = r0 + m, bb = g >> 12, nn = g & 4095;
        int kcol = k0 + kq, h = kcol >> 6, d = kcol & 63;
        float4 f = *(const float4*)&inter[(((size_t)bb * Hn + h) * Nn + nn) * DHn + d];
        *(uint2*)&As[m][kq] = f2bf_pack4(f);
      }
#pragma unroll
      for (int i = 0; i < 4; ++i) {
        int e = tid + i * 128;
        int k = e >> 4, nq = (e & 15) * 4;
        float4 f = *(const float4*)&Wo[(size_t)(k0 + k) * Dn + c0 + nq];
        Bs[nq + 0][k] = f2bf(f.x); Bs[nq + 1][k] = f2bf(f.y);
        Bs[nq + 2][k] = f2bf(f.z); Bs[nq + 3][k] = f2bf(f.w);
      }
    }
    __syncthreads();
    v16bf a = ld_frag<false>(&As[mloc][0], hi);
    v16bf b = ld_frag<true>(&Bs[wave * 16 + mloc][0], hi);
    acc = wmma_bf16(a, b, acc);
  }
  const int col = c0 + wave * 16 + mloc;
  const float bias = bo[col];
#pragma unroll
  for (int r = 0; r < 8; ++r) {
    int g = r0 + r + 8 * hi;
    outf[(size_t)g * Dn + col] = acc[r] + bias;
  }
}

// ================= host launcher =================
extern "C" void kernel_launch(void* const* d_in, const int* in_sizes, int n_in,
                              void* d_out, int out_size, void* d_ws,
                              size_t ws_size, hipStream_t stream) {
  const float* x  = (const float*)d_in[0];
  const float* Wq = (const float*)d_in[1];
  const float* Wk = (const float*)d_in[2];
  const float* Wv = (const float*)d_in[3];
  const float* Wo = (const float*)d_in[4];
  const float* bo = (const float*)d_in[5];

  float* out   = (float*)d_out;
  float* fo    = out;
  float* qf    = fo + SZ_FO;
  float* kf    = qf + SZ_QKV;
  float* vf    = kf + SZ_QKV;
  float* inter = vf + SZ_QKV;

  unsigned short* ws  = (unsigned short*)d_ws;
  unsigned short* qbf = ws;
  unsigned short* kbf = qbf + SZ_QKV;
  unsigned short* vbf = kbf + SZ_QKV;
  unsigned short* ibf = vbf + SZ_QKV;
  unsigned short* xbf = ibf + SZ_QKV;
  unsigned short* wqt = xbf + (size_t)Mrows * Dn;
  unsigned short* wkt = wqt + SZ_W;
  unsigned short* wvt = wkt + SZ_W;
  unsigned short* wot = wvt + SZ_W;
  const size_t need = 2ull * (4 * SZ_QKV + (size_t)Mrows * Dn + 4 * SZ_W);
  const bool pre = (d_ws != nullptr) && (ws_size >= need);

  if (pre) {
    const int n4 = (int)((size_t)Mrows * Dn / 4);
    conv_f2bf_kernel<<<(n4 + 255) / 256, 256, 0, stream>>>(x, xbf, n4);
    const int nw = (int)SZ_W;
    transpose_w_kernel<<<(nw + 255) / 256, 256, 0, stream>>>(Wq, wqt, Dn, INNERn);
    transpose_w_kernel<<<(nw + 255) / 256, 256, 0, stream>>>(Wk, wkt, Dn, INNERn);
    transpose_w_kernel<<<(nw + 255) / 256, 256, 0, stream>>>(Wv, wvt, Dn, INNERn);
    transpose_w_kernel<<<(nw + 255) / 256, 256, 0, stream>>>(Wo, wot, INNERn, Dn);
    proj_kernel<true><<<dim3(Mrows / 16, 24), dim3(128), 0, stream>>>(
        x, Wq, Wk, Wv, xbf, wqt, wkt, wvt, qf, kf, vf, qbf, kbf, vbf);
    attn_kernel<true><<<dim3(Nn / 64, BHn), dim3(128), 0, stream>>>(
        qf, kf, vf, qbf, kbf, vbf, inter, ibf);
    outproj_kernel<true><<<dim3(Mrows / 16, Dn / 64), dim3(128), 0, stream>>>(
        inter, ibf, Wo, wot, bo, fo);
  } else {
    proj_kernel<false><<<dim3(Mrows / 16, 24), dim3(128), 0, stream>>>(
        x, Wq, Wk, Wv, xbf, wqt, wkt, wvt, qf, kf, vf, qbf, kbf, vbf);
    attn_kernel<false><<<dim3(Nn / 64, BHn), dim3(128), 0, stream>>>(
        qf, kf, vf, qbf, kbf, vbf, inter, ibf);
    outproj_kernel<false><<<dim3(Mrows / 16, Dn / 64), dim3(128), 0, stream>>>(
        inter, ibf, Wo, wot, bo, fo);
  }
}
